// HANModel_25005299598065
// MI455X (gfx1250) — matrix operand order repaired and checked by reading, not compiled, not run on hardware
//
#include <hip/hip_runtime.h>

// ---------------------------------------------------------------------------
// HAN forward for MI455X (gfx1250, wave32, WMMA bf16 w/ fp32 accumulate)
// S=64 sentences, W=128 words, D=256, HW=256, HS=512, C=13
// B weights pre-swizzled into WMMA fragment order (2x b128 per fragment);
// recurrent hidden state kept fp32 in LDS with an A-fragment-swizzled bf16
// shadow (2x ds_load_b128 per fragment).
// ---------------------------------------------------------------------------

typedef __attribute__((ext_vector_type(16))) __bf16 bf16x16;
typedef __attribute__((ext_vector_type(8)))  float  f32x8;

union Frag { unsigned u[8]; bf16x16 v; };

__device__ __forceinline__ unsigned f2bf1(float f) {
  unsigned u = __float_as_uint(f);
  return (u + 0x7FFFu + ((u >> 16) & 1u)) >> 16;   // round-to-nearest-even
}
__device__ __forceinline__ unsigned packbf(float lo, float hi) {
  return f2bf1(lo) | (f2bf1(hi) << 16);
}
__device__ __forceinline__ float sigmoidf_(float x) { return 1.0f / (1.0f + __expf(-x)); }

__device__ __forceinline__ f32x8 wmma_bf16(const Frag& a, const Frag& b, f32x8 c) {
  return __builtin_amdgcn_wmma_f32_16x16x32_bf16(false, a.v, false, b.v, (short)0, c, false, false);
}

// A fragment (16x32 bf16, MxK) from row-major bf16 rows (uint pairs):
// k = kc*32 + (v>=4)*16 + (lane>=16)*8 + (v&3)*2  -> two 16B-contiguous groups
__device__ __forceinline__ void load_a_frag(Frag& f, const unsigned* __restrict__ row,
                                            int kc, int lane) {
  int kb = kc * 16 + ((lane >> 4) << 2);
#pragma unroll
  for (int v = 0; v < 8; ++v)
    f.u[v] = row[kb + ((v >> 2) << 3) + (v & 3)];
}

// Fragment-swizzled load: 8 consecutive uints per lane (32B) -> 2x b128.
__device__ __forceinline__ void load_frag_swz(Frag& f, const unsigned* __restrict__ fragBase,
                                              int lane) {
  const unsigned* p = fragBase + lane * 8;
#pragma unroll
  for (int v = 0; v < 8; ++v) f.u[v] = p[v];
}

// ---------------------------------------------------------------------------
// 1) Pack weights into WMMA B-fragment order.
//    Fragment index: ((nt*KC + kc)*32 + lane)*8 + v
//    element: n = nt*16 + (lane&15), k = kc*32 + (v>=4)*16 + (lane>=16)*8 + (v&3)*2  (pair k,k+1)
//    WiP/WhP: N=768 (NT=48), K=256 (KC=8), per dir 98304 uints.
//    waP:     N=512 (NT=32), K=512 (KC=16), 131072 uints.
__global__ void prep_kernel(const float* __restrict__ wWi, const float* __restrict__ wWh,
                            const float* __restrict__ waW,
                            unsigned* __restrict__ WiP, unsigned* __restrict__ WhP,
                            unsigned* __restrict__ waP) {
  int id = blockIdx.x * 256 + threadIdx.x;               // 524288 total
  if (id < 393216) {                                     // WiP then WhP
    const float* W = (id < 196608) ? wWi : wWh;
    unsigned* P    = (id < 196608) ? WiP : WhP;
    int j = (id < 196608) ? id : id - 196608;
    int d = j / 98304, r = j % 98304;
    int v = r & 7, lane = (r >> 3) & 31, fc = r >> 8;    // fc = nt*8 + kc
    int kc = fc & 7, nt = fc >> 3;
    int n = nt * 16 + (lane & 15);
    int k = kc * 32 + ((v >> 2) << 4) + ((lane >> 4) << 3) + (v & 3) * 2;
    const float* s = W + ((long)d * 768 + n) * 256 + k;
    P[j] = packbf(s[0], s[1]);
  } else {                                               // waP
    int j = id - 393216;
    int v = j & 7, lane = (j >> 3) & 31, fc = j >> 8;    // fc = nt*16 + kc
    int kc = fc & 15, nt = fc >> 4;
    int n = nt * 16 + (lane & 15);
    int k = kc * 32 + ((v >> 2) << 4) + ((lane >> 4) << 3) + (v & 3) * 2;
    const float* s = waW + n * 512 + k;
    waP[j] = packbf(s[0], s[1]);
  }
}

// ---------------------------------------------------------------------------
// 2) Gather embeddings -> Xbf[r=w*64+s][256] bf16 (rows = word-major).
__global__ void gather_kernel(const int* __restrict__ doc, const float* __restrict__ emb,
                              unsigned* __restrict__ Xbf) {
  int r = blockIdx.x;                  // 0..8191
  int s = r & 63, w = r >> 6;
  long tok = doc[s * 128 + w];
  const float* e = emb + tok * 256;
  int k = threadIdx.x * 4;             // 64 threads/row
  float4 f = *(const float4*)(e + k);
  Xbf[r * 128 + (k >> 1)]     = packbf(f.x, f.y);
  Xbf[r * 128 + (k >> 1) + 1] = packbf(f.z, f.w);
}

// ---------------------------------------------------------------------------
// 3) GI[d][r][n] = Xbf[r,:] . WiT[d][:,n] + bi   (M=8192, N=768, K=256, 2 dirs)
__global__ void gi_gemm_kernel(const unsigned* __restrict__ Xbf, const unsigned* __restrict__ WiP,
                               const float* __restrict__ wbi, float* __restrict__ GI) {
  int lane = threadIdx.x & 31, wave = threadIdx.x >> 5;
  int tile = blockIdx.x * 8 + wave;                      // 49152 tiles
  int d = tile / (512 * 48), rm = tile % (512 * 48);
  int mt = rm / 48, nt = rm % 48;
  int m0 = mt * 16;
  int col = nt * 16 + (lane & 15);
  const unsigned* arow = Xbf + (m0 + (lane & 15)) * 128;
  const unsigned* bbase = WiP + d * 98304 + nt * 2048;   // + kc*256
  f32x8 acc = {};
#pragma unroll
  for (int kc = 0; kc < 8; ++kc) {
    Frag a, b;
    load_a_frag(a, arow, kc, lane);
    load_frag_swz(b, bbase + kc * 256, lane);
    acc = wmma_bf16(a, b, acc);
  }
  float bias = wbi[d * 768 + col];
  int rbase = (lane >> 4) << 3;
#pragma unroll
  for (int c = 0; c < 8; ++c)
    GI[(d * 8192 + m0 + c + rbase) * 768 + col] = acc[c] + bias;
}

// ---------------------------------------------------------------------------
// 4) Sequential word GRU, one workgroup per direction (32 waves on one WGP).
//    h fp32 in LDS + A-fragment-swizzled bf16 shadow; each wave owns 2
//    gate-triples so all three gates of a 16x16 (s,j) block stay in registers.
__global__ void __launch_bounds__(1024) word_gru_kernel(
    const float* __restrict__ GI, const unsigned* __restrict__ WhP,
    const float* __restrict__ wbh, unsigned short* __restrict__ wordenc) {
  extern __shared__ char smem[];
  float*    h_f   = (float*)smem;                        // [64][260] fp32 (padded)
  unsigned* h_swz = (unsigned*)(smem + 64 * 260 * 4);    // A-frag swizzled bf16: 4*2048 uints
  unsigned short* h_swz16 = (unsigned short*)h_swz;

  const int d = blockIdx.x;
  const int tid = threadIdx.x, lane = tid & 31, wave = tid >> 5;
  const int t0 = wave * 2;                               // 64 triples total

  for (int i = tid; i < 64 * 260; i += 1024) h_f[i] = 0.0f;
  for (int i = tid; i < 8192; i += 1024) h_swz[i] = 0;
  __syncthreads();

  for (int w = 0; w < 128; ++w) {
    f32x8 acc[2][3] = {};
#pragma unroll
    for (int tau = 0; tau < 2; ++tau) {
      int t = t0 + tau;
      int mt = t >> 4, jt = t & 15;
      const unsigned* afrag = h_swz + mt * 2048;         // + kc*256
      const unsigned* bbase = WhP + d * 98304 + jt * 2048; // gate g adds g*16*2048... see below
#pragma unroll
      for (int kc = 0; kc < 8; ++kc) {
        Frag a;
        load_frag_swz(a, afrag + kc * 256, lane);
#pragma unroll
        for (int g = 0; g < 3; ++g) {
          Frag b;
          // column tile for gate g is (g*16 + jt) -> frag (g*16+jt)*8 + kc
          load_frag_swz(b, bbase + g * 32768 + kc * 256, lane);
          acc[tau][g] = wmma_bf16(a, b, acc[tau][g]);
        }
      }
    }
    __syncthreads();   // all reads of h (state w-1) complete

#pragma unroll
    for (int tau = 0; tau < 2; ++tau) {
      int t = t0 + tau;
      int mt = t >> 4, jt = t & 15;
      int n = jt * 16 + (lane & 15);
      int mb = mt * 16 + ((lane >> 4) << 3);
      float bhr = wbh[d * 768 + n];
      float bhz = wbh[d * 768 + 256 + n];
      float bhn = wbh[d * 768 + 512 + n];
      // inverse A-frag mapping pieces that depend only on n
      int kcp = n >> 5, kk = n & 31;
      int laneLo = (((kk & 15) >= 8) ? 16 : 0);
      int vp = ((kk >= 16) ? 4 : 0) + ((kk & 7) >> 1);
      int half = kk & 1;
#pragma unroll
      for (int c = 0; c < 8; ++c) {
        int m = mb + c;
        const float* gi = GI + (d * 8192 + w * 64 + m) * 768 + n;
        float r  = sigmoidf_(gi[0]   + acc[tau][0][c] + bhr);
        float z  = sigmoidf_(gi[256] + acc[tau][1][c] + bhz);
        float nn = tanhf   (gi[512] + r * (acc[tau][2][c] + bhn));
        float hold = h_f[m * 260 + n];
        float hnew = (1.0f - z) * nn + z * hold;
        h_f[m * 260 + n] = hnew;
        unsigned hb = f2bf1(hnew);
        // write bf16 into A-fragment-swizzled shadow
        int mtp = m >> 4, lm = m & 15;
        int uidx = mtp * 2048 + kcp * 256 + (lm + laneLo) * 8 + vp;
        h_swz16[uidx * 2 + half] = (unsigned short)hb;
        wordenc[(w * 64 + m) * 512 + d * 256 + n] = (unsigned short)hb;
      }
    }
    __syncthreads();   // state w published
  }
}

// ---------------------------------------------------------------------------
// 5) u_word = tanh(wordenc @ waT + b)   (M=8192, N=512, K=512)
__global__ void attn_gemm_kernel(const unsigned* __restrict__ Wenc, const unsigned* __restrict__ waP,
                                 const float* __restrict__ wab, float* __restrict__ u_word) {
  int lane = threadIdx.x & 31, wave = threadIdx.x >> 5;
  int tile = blockIdx.x * 8 + wave;                      // 16384 tiles
  int mt = tile / 32, nt = tile % 32;
  int col = nt * 16 + (lane & 15);
  const unsigned* arow = Wenc + (mt * 16 + (lane & 15)) * 256;
  const unsigned* bbase = waP + nt * 4096;               // + kc*256 (KC=16)
  f32x8 acc = {};
#pragma unroll
  for (int kc = 0; kc < 16; ++kc) {
    Frag a, b;
    load_a_frag(a, arow, kc, lane);
    load_frag_swz(b, bbase + kc * 256, lane);
    acc = wmma_bf16(a, b, acc);
  }
  float bias = wab[col];
  int rbase = (lane >> 4) << 3;
#pragma unroll
  for (int c = 0; c < 8; ++c)
    u_word[(mt * 16 + c + rbase) * 512 + col] = tanhf(acc[c] + bias);
}

// ---------------------------------------------------------------------------
// 6) sent_summ[s][n] = sum_w u_word[w*64+s][n]   (word_w == 1)
__global__ void wordsum_kernel(const float* __restrict__ u_word, float* __restrict__ ssum) {
  int id = blockIdx.x * 256 + threadIdx.x;               // 32768
  int s = id >> 9, n = id & 511;
  float a = 0.0f;
  for (int w = 0; w < 128; ++w) a += u_word[((w << 6) + s) * 512 + n];
  ssum[id] = a;
}

// ---------------------------------------------------------------------------
// 7) Degenerate sentence bi-GRU (zero hidden state).
__global__ void sentgru_kernel(const float* __restrict__ ssum, const float* __restrict__ sWi,
                               const float* __restrict__ sbi, const float* __restrict__ sbh,
                               float* __restrict__ sentenc) {
  int id = blockIdx.x * 256 + threadIdx.x;               // 65536
  int d = id >> 15, rm = id & 32767;
  int s = rm >> 9, j = rm & 511;
  const float* x = ssum + s * 512;
  float g[3];
#pragma unroll
  for (int gg = 0; gg < 3; ++gg) {
    const float* wr = sWi + ((long)d * 1536 + gg * 512 + j) * 512;
    float a = sbi[d * 1536 + gg * 512 + j];
    for (int k = 0; k < 512; ++k) a += x[k] * wr[k];
    g[gg] = a;
  }
  float hr = sbh[d * 1536 + j], hz = sbh[d * 1536 + 512 + j], hn = sbh[d * 1536 + 1024 + j];
  float r = sigmoidf_(g[0] + hr), z = sigmoidf_(g[1] + hz);
  float nn = tanhf(g[2] + r * hn);
  sentenc[s * 1024 + d * 512 + j] = (1.0f - z) * nn;
}

// 8) u_sent = tanh(sentenc @ sa_W.T + sa_b)
__global__ void usent_kernel(const float* __restrict__ sentenc, const float* __restrict__ saW,
                             const float* __restrict__ sab, float* __restrict__ usent) {
  int id = blockIdx.x * 256 + threadIdx.x;               // 65536
  int s = id >> 10, n = id & 1023;
  const float* x = sentenc + s * 1024;
  const float* wr = saW + n * 1024;
  float a = sab[n];
  for (int k = 0; k < 1024; ++k) a += x[k] * wr[k];
  usent[id] = tanhf(a);
}

// 9) word_w / sent_w are softmax over a size-1 axis -> all ones.
__global__ void ones_kernel(float* __restrict__ out) {
  int id = blockIdx.x * 256 + threadIdx.x;
  if (id < 8256) out[id] = 1.0f;
}

// 10) doc = sum_s u_sent; logits; log_softmax.
__global__ void __launch_bounds__(1024) final_kernel(const float* __restrict__ usent,
                                                     const float* __restrict__ doW,
                                                     const float* __restrict__ dob,
                                                     float* __restrict__ out) {
  __shared__ float doc[1024];
  __shared__ float lg[13];
  int t = threadIdx.x;
  float a = 0.0f;
  for (int s = 0; s < 64; ++s) a += usent[s * 1024 + t];
  doc[t] = a;
  __syncthreads();
  if (t < 13) {
    const float* wr = doW + t * 1024;
    float l = dob[t];
    for (int k = 0; k < 1024; ++k) l += doc[k] * wr[k];
    lg[t] = l;
  }
  __syncthreads();
  if (t == 0) {
    float m = lg[0];
    for (int i = 1; i < 13; ++i) m = fmaxf(m, lg[i]);
    float se = 0.0f;
    for (int i = 0; i < 13; ++i) se += __expf(lg[i] - m);
    float lse = m + logf(se);
    for (int i = 0; i < 13; ++i) out[8256 + i] = lg[i] - lse;
  }
}

// ---------------------------------------------------------------------------
// Workspace layout (bytes)
static const size_t OFF_XBF   = 0;                         // 8192*128 u32  = 4 MB
static const size_t OFF_WIP   = OFF_XBF   + 8192u*128*4;   // 196608 u32
static const size_t OFF_WHP   = OFF_WIP   + 196608u*4;
static const size_t OFF_WAP   = OFF_WHP   + 196608u*4;     // 131072 u32
static const size_t OFF_GI    = OFF_WAP   + 131072u*4;     // 2*8192*768 f32 = 50.3 MB
static const size_t OFF_WENC  = OFF_GI    + 2u*8192*768*4; // 8192*512 bf16
static const size_t OFF_UWORD = OFF_WENC  + 8192u*512*2;   // 8192*512 f32
static const size_t OFF_SSUM  = OFF_UWORD + 8192u*512*4;   // 64*512 f32
static const size_t OFF_SENC  = OFF_SSUM  + 64u*512*4;     // 64*1024 f32
static const size_t OFF_USENT = OFF_SENC  + 64u*1024*4;    // 64*1024 f32

extern "C" void kernel_launch(void* const* d_in, const int* in_sizes, int n_in,
                              void* d_out, int out_size, void* d_ws, size_t ws_size,
                              hipStream_t stream) {
  const int*   doc = (const int*)  d_in[0];
  const float* emb = (const float*)d_in[1];
  const float* wWi = (const float*)d_in[2];
  const float* wWh = (const float*)d_in[3];
  const float* wbi = (const float*)d_in[4];
  const float* wbh = (const float*)d_in[5];
  const float* sWi = (const float*)d_in[6];
  const float* sbi = (const float*)d_in[8];
  const float* sbh = (const float*)d_in[9];
  const float* waW = (const float*)d_in[10];
  const float* wab = (const float*)d_in[11];
  const float* saW = (const float*)d_in[13];
  const float* sab = (const float*)d_in[14];
  const float* doW = (const float*)d_in[16];
  const float* dob = (const float*)d_in[17];
  float* out = (float*)d_out;
  char*  ws  = (char*)d_ws;

  unsigned*       Xbf   = (unsigned*)      (ws + OFF_XBF);
  unsigned*       WiP   = (unsigned*)      (ws + OFF_WIP);
  unsigned*       WhP   = (unsigned*)      (ws + OFF_WHP);
  unsigned*       waP   = (unsigned*)      (ws + OFF_WAP);
  float*          GI    = (float*)         (ws + OFF_GI);
  unsigned short* Wenc  = (unsigned short*)(ws + OFF_WENC);
  float*          uword = (float*)         (ws + OFF_UWORD);
  float*          ssum  = (float*)         (ws + OFF_SSUM);
  float*          senc  = (float*)         (ws + OFF_SENC);
  float*          usent = (float*)         (ws + OFF_USENT);

  prep_kernel  <<<2048, 256, 0, stream>>>(wWi, wWh, waW, WiP, WhP, waP);
  gather_kernel<<<8192,  64, 0, stream>>>(doc, emb, Xbf);
  gi_gemm_kernel<<<6144, 256, 0, stream>>>(Xbf, WiP, wbi, GI);

  const unsigned ldsBytes = 64u * 260 * 4 + 8192u * 4;     // 97.0 KB < 320 KB/WGP
  word_gru_kernel<<<2, 1024, ldsBytes, stream>>>(GI, WhP, wbh, Wenc);

  attn_gemm_kernel<<<2048, 256, 0, stream>>>((const unsigned*)Wenc, waP, wab, uword);
  wordsum_kernel <<<128,  256, 0, stream>>>(uword, ssum);
  sentgru_kernel <<<256,  256, 0, stream>>>(ssum, sWi, sbi, sbh, senc);
  usent_kernel   <<<256,  256, 0, stream>>>(senc, saW, sab, usent);
  ones_kernel    <<<33,   256, 0, stream>>>(out);
  final_kernel   <<<1,   1024, 0, stream>>>(usent, doW, dob, out);
}
